// CausalAttention_18425409699889
// MI455X (gfx1250) — matrix operand order repaired and checked by reading, not compiled, not run on hardware
//
#include <hip/hip_runtime.h>

#define TSEQ 2048
#define EMB  1024
#define NH   16
#define HD   64
#define NB   2
#define NTOK (NB * TSEQ)   // 4096
#define KDIM EMB

typedef __attribute__((ext_vector_type(16))) __bf16 v16bf;
typedef __attribute__((ext_vector_type(8)))  __bf16 v8bf;
typedef __attribute__((ext_vector_type(8)))  float  v8f;

__device__ inline v8f wmma_bf16(v16bf a, v16bf b, v8f c) {
  // D = A(16x32) * B(32x16) + C, f32 accumulate
  return __builtin_amdgcn_wmma_f32_16x16x32_bf16(false, a, false, b, (short)0, c, false, false);
}

__device__ inline v16bf cat16(v8bf lo, v8bf hi) {
  v16bf r;
#pragma unroll
  for (int i = 0; i < 8; ++i) { r[i] = lo[i]; r[i + 8] = hi[i]; }
  return r;
}

// ---------------------------------------------------------------------------
// f32 -> bf16 conversion (x and weights), 4 elements/thread
// ---------------------------------------------------------------------------
__global__ __launch_bounds__(256) void cvt_kernel(const float* __restrict__ s,
                                                  __bf16* __restrict__ d, int n) {
  int i = (blockIdx.x * 256 + threadIdx.x) * 4;
  if (i >= n) return;
  float4 f = *(const float4*)(s + i);
  union { __bf16 h[4]; uint2 u; } pk;
  pk.h[0] = (__bf16)f.x; pk.h[1] = (__bf16)f.y;
  pk.h[2] = (__bf16)f.z; pk.h[3] = (__bf16)f.w;
  *(uint2*)(d + i) = pk.u;
}

// ---------------------------------------------------------------------------
// GEMM: out[m,n] = sum_k A[m,k] * W[n,k] + bias[n]
// M=4096, N=K=1024. Block = 8 waves (4Mx2N), 128x128 tile; wave = 32(M)x64(N).
// Per K-step: 8 loads -> 8 WMMA.
// OUT_MODE 0: bf16 to [B,H,T,D]   (q,k projections)
// OUT_MODE 1: bf16 to [B,H,D,T]   (v projection, transposed for attention)
// OUT_MODE 2: f32  to [M,E]       (output projection -> d_out)
// ---------------------------------------------------------------------------
template <int OUT_MODE>
__global__ __launch_bounds__(256) void gemm_kernel(const __bf16* __restrict__ A,
                                                   const __bf16* __restrict__ Wb,
                                                   const float* __restrict__ bias,
                                                   void* __restrict__ outp) {
  const int lane = threadIdx.x & 31;
  const int w    = threadIdx.x >> 5;
  const int g    = lane >> 4;     // half-wave group
  const int ln   = lane & 15;
  const int mw   = blockIdx.y * 128 + (w & 3) * 32;
  const int nw   = blockIdx.x * 128 + (w >> 2) * 64;

  const __bf16* a0p = A + (size_t)(mw + ln) * KDIM;
  const __bf16* a1p = A + (size_t)(mw + 16 + ln) * KDIM;
  const __bf16* bp0 = Wb + (size_t)(nw + ln) * KDIM + 16 * g;
  const __bf16* bp1 = Wb + (size_t)(nw + 16 + ln) * KDIM + 16 * g;
  const __bf16* bp2 = Wb + (size_t)(nw + 32 + ln) * KDIM + 16 * g;
  const __bf16* bp3 = Wb + (size_t)(nw + 48 + ln) * KDIM + 16 * g;
  const int ao0 = 16 * g, ao1 = 16 * g + 8;

  v8f c[2][4] = {};
#pragma unroll 2
  for (int k0 = 0; k0 < KDIM; k0 += 32) {
    __builtin_prefetch(a0p + k0 + 256, 0, 0);   // global_prefetch_b8, ~8 iters ahead
    __builtin_prefetch(a1p + k0 + 256, 0, 0);
    v16bf a0 = cat16(*(const v8bf*)(a0p + k0 + ao0), *(const v8bf*)(a0p + k0 + ao1));
    v16bf a1 = cat16(*(const v8bf*)(a1p + k0 + ao0), *(const v8bf*)(a1p + k0 + ao1));
    v16bf b0 = *(const v16bf*)(bp0 + k0);
    v16bf b1 = *(const v16bf*)(bp1 + k0);
    v16bf b2 = *(const v16bf*)(bp2 + k0);
    v16bf b3 = *(const v16bf*)(bp3 + k0);
    c[0][0] = wmma_bf16(a0, b0, c[0][0]);
    c[0][1] = wmma_bf16(a0, b1, c[0][1]);
    c[0][2] = wmma_bf16(a0, b2, c[0][2]);
    c[0][3] = wmma_bf16(a0, b3, c[0][3]);
    c[1][0] = wmma_bf16(a1, b0, c[1][0]);
    c[1][1] = wmma_bf16(a1, b1, c[1][1]);
    c[1][2] = wmma_bf16(a1, b2, c[1][2]);
    c[1][3] = wmma_bf16(a1, b3, c[1][3]);
  }
  float bv[4];
#pragma unroll
  for (int u = 0; u < 4; ++u) bv[u] = bias[nw + 16 * u + ln];
#pragma unroll
  for (int t = 0; t < 2; ++t)
#pragma unroll
    for (int u = 0; u < 4; ++u)
#pragma unroll
      for (int r = 0; r < 8; ++r) c[t][u][r] += bv[u];

  if (OUT_MODE == 2) {
    float* out = (float*)outp;
#pragma unroll
    for (int t = 0; t < 2; ++t)
#pragma unroll
      for (int r = 0; r < 8; ++r) {
        const int m = mw + 16 * t + 8 * g + r;
#pragma unroll
        for (int u = 0; u < 4; ++u)
          out[(size_t)m * EMB + nw + 16 * u + ln] = c[t][u][r];
      }
  } else if (OUT_MODE == 0) {
    __bf16* out = (__bf16*)outp;
#pragma unroll
    for (int t = 0; t < 2; ++t)
#pragma unroll
      for (int r = 0; r < 8; ++r) {
        const int m = mw + 16 * t + 8 * g + r;
        const int b = m >> 11, tt = m & (TSEQ - 1);
#pragma unroll
        for (int u = 0; u < 4; ++u) {
          const int n = nw + 16 * u + ln;
          out[((size_t)(b * NH + (n >> 6)) * TSEQ + tt) * HD + (n & 63)] =
              (__bf16)c[t][u][r];
        }
      }
  } else {  // OUT_MODE == 1: v transposed [B,H,D,T]; 8 consecutive tokens -> b128 store
    __bf16* out = (__bf16*)outp;
#pragma unroll
    for (int t = 0; t < 2; ++t) {
      const int m0 = mw + 16 * t + 8 * g;         // tokens m0..m0+7
      const int b = m0 >> 11, t0 = m0 & (TSEQ - 1);
#pragma unroll
      for (int u = 0; u < 4; ++u) {
        const int n = nw + 16 * u + ln;
        union { __bf16 h[8]; uint4 v; } pk;
#pragma unroll
        for (int r = 0; r < 8; ++r) pk.h[r] = (__bf16)c[t][u][r];
        *(uint4*)(out + ((size_t)(b * NH + (n >> 6)) * HD + (n & 63)) * TSEQ + t0) = pk.v;
      }
    }
  }
}

// ---------------------------------------------------------------------------
// Async-copy one V tile (rows d=0..63, cols j0..j0+31, bf16) from global
// [B,H,D,T] into a per-wave LDS buffer laid out [d][32]. 8 x B128 per lane,
// tracked by ASYNCcnt (no barriers needed: buffer is wave-private).
// ---------------------------------------------------------------------------
__device__ inline void async_v_tile(const __bf16* __restrict__ vh, int j0,
                                    unsigned lb, int lane) {
  const int part = lane & 3;            // which 16B chunk of a 64B row
  const int d0   = lane >> 2;           // rows d0, d0+8, ..., d0+56
  const __bf16* gp = vh + (size_t)d0 * TSEQ + j0 + part * 8;
  const unsigned lo = lb + (unsigned)(d0 * 64 + part * 16);
#pragma unroll
  for (int k = 0; k < 8; ++k) {
    asm volatile("global_load_async_to_lds_b128 %0, %1, off"
                 :: "v"(lo + (unsigned)(k * 512)),
                    "v"((unsigned long long)(uintptr_t)(gp + (size_t)k * 8 * TSEQ))
                 : "memory");
  }
}

// ---------------------------------------------------------------------------
// Async-copy one Q tile (rows j0..j0+31, 64 d-elements, bf16) from global
// [B,H,T,D] into a per-wave LDS buffer laid out [row][64].
// ---------------------------------------------------------------------------
__device__ inline void async_q_tile(const __bf16* __restrict__ qh, int j0,
                                    unsigned lb, int lane) {
  const int part = lane & 7;            // which 16B chunk of a 128B row
  const int r0   = lane >> 3;           // rows r0, r0+4, ..., r0+28
  const __bf16* gp = qh + (size_t)(j0 + r0) * HD + part * 8;
  const unsigned lo = lb + (unsigned)(r0 * 128 + part * 16);
#pragma unroll
  for (int k = 0; k < 8; ++k) {
    asm volatile("global_load_async_to_lds_b128 %0, %1, off"
                 :: "v"(lo + (unsigned)(k * 512)),          // 4 rows * 128B
                    "v"((unsigned long long)(uintptr_t)(gp + (size_t)k * 4 * HD))
                 : "memory");
  }
}

// ---------------------------------------------------------------------------
// Flash attention step over one 32-wide j-tile (S^T formulation).
// qbuf/vbuf must be STATIC LDS addresses (direct GEPs into __shared__) so the
// reads lower to ds_load (DScnt only), decoupled from the async stream.
// MASKED=false: fully-unmasked fast path (no cndmask/exec code at all).
// ---------------------------------------------------------------------------
template <bool MASKED>
__device__ inline void attn_step(const __bf16* qbuf,   // LDS [32][64]
                                 const __bf16* vbuf,   // LDS [64][32]
                                 int j0, int ibase, int g, int ln,
                                 const v16bf& Bk0, const v16bf& Bk1,
                                 v8f& O0, v8f& O1, v8f& O2, v8f& O3,
                                 float& m, float& l) {
  const float CEXP = 0.125f * 1.4426950408889634f;  // (1/sqrt(D)) * log2(e)
  // A operand: q rows (tile-local) from LDS
  const __bf16* q0 = qbuf + ln * HD;
  const __bf16* q1 = qbuf + (16 + ln) * HD;
  v16bf a00 = cat16(*(const v8bf*)(q0 + 16 * g),      *(const v8bf*)(q0 + 8 + 16 * g));
  v16bf a01 = cat16(*(const v8bf*)(q0 + 32 + 16 * g), *(const v8bf*)(q0 + 40 + 16 * g));
  v16bf a10 = cat16(*(const v8bf*)(q1 + 16 * g),      *(const v8bf*)(q1 + 8 + 16 * g));
  v16bf a11 = cat16(*(const v8bf*)(q1 + 32 + 16 * g), *(const v8bf*)(q1 + 40 + 16 * g));
  const v8f z = {};
  v8f st0 = wmma_bf16(a00, Bk0, z);
  st0     = wmma_bf16(a01, Bk1, st0);
  v8f st1 = wmma_bf16(a10, Bk0, z);
  st1     = wmma_bf16(a11, Bk1, st1);

  if (MASKED) {  // element (j,i) masked when j > i; raw-logit units
    const int ig = ibase + ln;
#pragma unroll
    for (int r = 0; r < 8; ++r) {
      if (j0 + r + 8 * g > ig)      st0[r] = -__builtin_inff();
      if (j0 + 16 + r + 8 * g > ig) st1[r] = -__builtin_inff();
    }
  }
  // online softmax stats; column i lives in lanes {ln, ln+16}
  float mloc = -__builtin_inff();
#pragma unroll
  for (int r = 0; r < 8; ++r) mloc = fmaxf(mloc, fmaxf(st0[r], st1[r]));
  mloc = fmaxf(mloc, __shfl_xor(mloc, 16, 32));
  const float mnew = fmaxf(m, mloc);

  v16bf P;                                   // lands lane-exact in A-operand layout
  float ps = 0.0f;
#pragma unroll
  for (int r = 0; r < 8; ++r) {
    float p0 = __builtin_amdgcn_exp2f((st0[r] - mnew) * CEXP);
    float p1 = __builtin_amdgcn_exp2f((st1[r] - mnew) * CEXP);
    ps += p0 + p1;
    P[r]     = (__bf16)p0;
    P[r + 8] = (__bf16)p1;
  }
  ps += __shfl_xor(ps, 16, 32);

  if (__any(mnew > m)) {                     // rare after warmup: skip rescale
    const float alpha = __builtin_amdgcn_exp2f((m - mnew) * CEXP);
#pragma unroll
    for (int r = 0; r < 8; ++r) {            // row r+8g needs alpha from lane r+8g
      const float ar = __shfl(alpha, r + 8 * g, 32);
      O0[r] *= ar; O1[r] *= ar; O2[r] *= ar; O3[r] *= ar;
    }
    l = l * alpha + ps;
  } else {
    l += ps;
  }
  m = mnew;

  // O += P * V   (B operand from LDS tile: row d, 32 j-elements)
  v16bf bv0 = cat16(*(const v8bf*)(vbuf + (0  + ln) * 32 + 8 * g), *(const v8bf*)(vbuf + (0  + ln) * 32 + 16 + 8 * g));
  v16bf bv1 = cat16(*(const v8bf*)(vbuf + (16 + ln) * 32 + 8 * g), *(const v8bf*)(vbuf + (16 + ln) * 32 + 16 + 8 * g));
  v16bf bv2 = cat16(*(const v8bf*)(vbuf + (32 + ln) * 32 + 8 * g), *(const v8bf*)(vbuf + (32 + ln) * 32 + 16 + 8 * g));
  v16bf bv3 = cat16(*(const v8bf*)(vbuf + (48 + ln) * 32 + 8 * g), *(const v8bf*)(vbuf + (48 + ln) * 32 + 16 + 8 * g));
  O0 = wmma_bf16(P, bv0, O0);
  O1 = wmma_bf16(P, bv1, O1);
  O2 = wmma_bf16(P, bv2, O2);
  O3 = wmma_bf16(P, bv3, O3);
}

// ---------------------------------------------------------------------------
// Flash attention (reference semantics: rows i = key positions, softmax over
// query positions j <= i, values gathered at j).
// qb: q-projection [B,H,T,D]  (async-staged to LDS, A operand, j rows)
// kb: k-projection [B,H,T,D]  (transposed B operand, i rows)
// vt: v-projection [B,H,D,T]  (async-staged to LDS, B operand for P*V)
// ob: attention output [B,T,E] bf16
// Pipeline unrolled x2 so each step uses a compile-time buffer parity:
// tile t (t=0..nfull; tile nfull is the masked one) lives in buffer t&1.
// ---------------------------------------------------------------------------
__global__ __launch_bounds__(256) void attn_kernel(const __bf16* __restrict__ qb,
                                                   const __bf16* __restrict__ kb,
                                                   const __bf16* __restrict__ vt,
                                                   __bf16* __restrict__ ob) {
  __shared__ __bf16 vsm[2][8][HD * 32];          // double-buffered V tiles, per wave
  __shared__ __bf16 qsm[2][8][32 * HD];          // double-buffered Q tiles, per wave
  const int lane = threadIdx.x & 31;
  const int w    = threadIdx.x >> 5;
  const int g    = lane >> 4;
  const int ln   = lane & 15;
  const int bh   = blockIdx.y;
  const int b    = bh >> 4, h = bh & 15;
  const int ibase = blockIdx.x * 128 + w * 16;   // this wave's 16 key rows

  const __bf16* qh = qb + (size_t)bh * TSEQ * HD;
  const __bf16* kh = kb + (size_t)bh * TSEQ * HD;
  const __bf16* vh = vt + (size_t)bh * HD * TSEQ;

  const __bf16* q0b = &qsm[0][w][0];
  const __bf16* q1b = &qsm[1][w][0];
  const __bf16* v0b = &vsm[0][w][0];
  const __bf16* v1b = &vsm[1][w][0];
  const unsigned q0l = (unsigned)(uintptr_t)q0b, q1l = (unsigned)(uintptr_t)q1b;
  const unsigned v0l = (unsigned)(uintptr_t)v0b, v1l = (unsigned)(uintptr_t)v1b;

  // Persistent B operand: k rows i, transposed. Frags over d = [0,32), [32,64)
  const v16bf Bk0 = *(const v16bf*)(kh + (size_t)(ibase + ln) * HD + 0  + 16 * g);
  const v16bf Bk1 = *(const v16bf*)(kh + (size_t)(ibase + ln) * HD + 32 + 16 * g);

  v8f O0 = {}, O1 = {}, O2 = {}, O3 = {};
  float m = -__builtin_inff(), l = 0.0f;

  const int nfull = ibase >> 5;                  // tiles 0..nfull-1 full, nfull masked
  async_v_tile(vh, 0, v0l, lane);                // prologue: tile 0 -> buffer 0
  async_q_tile(qh, 0, q0l, lane);

  int t = 0;
  while (t + 2 <= nfull) {                       // two full tiles per trip
    async_v_tile(vh, 32 * (t + 1), v1l, lane);
    async_q_tile(qh, 32 * (t + 1), q1l, lane);
    asm volatile("s_wait_asynccnt 16" ::: "memory");
    attn_step<false>(q0b, v0b, 32 * t, ibase, g, ln, Bk0, Bk1, O0, O1, O2, O3, m, l);
    async_v_tile(vh, 32 * (t + 2), v0l, lane);
    async_q_tile(qh, 32 * (t + 2), q0l, lane);
    asm volatile("s_wait_asynccnt 16" ::: "memory");
    attn_step<false>(q1b, v1b, 32 * (t + 1), ibase, g, ln, Bk0, Bk1, O0, O1, O2, O3, m, l);
    t += 2;
  }
  if (t < nfull) {                               // one leftover full tile (in buffer 0)
    async_v_tile(vh, 32 * (t + 1), v1l, lane);
    async_q_tile(qh, 32 * (t + 1), q1l, lane);
    asm volatile("s_wait_asynccnt 16" ::: "memory");
    attn_step<false>(q0b, v0b, 32 * t, ibase, g, ln, Bk0, Bk1, O0, O1, O2, O3, m, l);
    t += 1;
    asm volatile("s_wait_asynccnt 0" ::: "memory");
    attn_step<true>(q1b, v1b, 32 * t, ibase, g, ln, Bk0, Bk1, O0, O1, O2, O3, m, l);
  } else {                                       // masked tile sits in buffer 0
    asm volatile("s_wait_asynccnt 0" ::: "memory");
    attn_step<true>(q0b, v0b, 32 * t, ibase, g, ln, Bk0, Bk1, O0, O1, O2, O3, m, l);
  }

  const float linv = 1.0f / l;
#pragma unroll
  for (int r = 0; r < 8; ++r) {
    const float lr = __shfl(linv, r + 8 * g, 32);
    const int i = ibase + r + 8 * g;
    __bf16* orow = ob + ((size_t)b * TSEQ + i) * EMB + h * HD;
    orow[0  + ln] = (__bf16)(O0[r] * lr);
    orow[16 + ln] = (__bf16)(O1[r] * lr);
    orow[32 + ln] = (__bf16)(O2[r] * lr);
    orow[48 + ln] = (__bf16)(O3[r] * lr);
  }
}

// ---------------------------------------------------------------------------
extern "C" void kernel_launch(void* const* d_in, const int* in_sizes, int n_in,
                              void* d_out, int out_size, void* d_ws, size_t ws_size,
                              hipStream_t stream) {
  const float* x  = (const float*)d_in[0];
  const float* Wq = (const float*)d_in[1];
  const float* bq = (const float*)d_in[2];
  const float* Wk = (const float*)d_in[3];
  const float* bk = (const float*)d_in[4];
  const float* Wv = (const float*)d_in[5];
  const float* bv = (const float*)d_in[6];
  const float* Wp = (const float*)d_in[7];
  const float* bp = (const float*)d_in[8];

  const size_t NTE = (size_t)NTOK * EMB;   // 4,194,304
  const size_t WSZ = (size_t)EMB * EMB;    // 1,048,576
  __bf16* xb   = (__bf16*)d_ws;            // ~50 MB total workspace use
  __bf16* qbuf = xb + NTE;
  __bf16* kbuf = qbuf + NTE;
  __bf16* vtb  = kbuf + NTE;
  __bf16* atb  = vtb + NTE;
  __bf16* wqb  = atb + NTE;
  __bf16* wkb  = wqb + WSZ;
  __bf16* wvb  = wkb + WSZ;
  __bf16* wpb  = wvb + WSZ;

  cvt_kernel<<<(int)(NTE / 1024), 256, 0, stream>>>(x,  xb,  (int)NTE);
  cvt_kernel<<<(int)(WSZ / 1024), 256, 0, stream>>>(Wq, wqb, (int)WSZ);
  cvt_kernel<<<(int)(WSZ / 1024), 256, 0, stream>>>(Wk, wkb, (int)WSZ);
  cvt_kernel<<<(int)(WSZ / 1024), 256, 0, stream>>>(Wv, wvb, (int)WSZ);
  cvt_kernel<<<(int)(WSZ / 1024), 256, 0, stream>>>(Wp, wpb, (int)WSZ);

  dim3 gg(EMB / 128, NTOK / 128);          // (8, 32)
  gemm_kernel<0><<<gg, 256, 0, stream>>>(xb, wqb, bq, qbuf);
  gemm_kernel<0><<<gg, 256, 0, stream>>>(xb, wkb, bk, kbuf);
  gemm_kernel<1><<<gg, 256, 0, stream>>>(xb, wvb, bv, vtb);

  dim3 ga(TSEQ / 128, NB * NH);            // (16, 32)
  attn_kernel<<<ga, 256, 0, stream>>>(qbuf, kbuf, vtb, atb);

  gemm_kernel<2><<<gg, 256, 0, stream>>>(atb, wpb, bp, (float*)d_out);
}
